// MaskedGraphAttentionLayer_3573412790307
// MI455X (gfx1250) — matrix operand order repaired and checked by reading, not compiled, not run on hardware
//
#include <hip/hip_runtime.h>
#include <math.h>

typedef _Float16 h8   __attribute__((ext_vector_type(8)));
typedef _Float16 v16h __attribute__((ext_vector_type(16)));
typedef float    v8f  __attribute__((ext_vector_type(8)));

union ABFrag { v16h v; h8 h[2]; };

#define IN_DIM  128
#define NCOEF   8
#define KDIM    1152          // 128 base + 128*8 spline features
#define BATCH   4
#define NTOK    2048
#define ROWS    (BATCH * NTOK)   // 8192
#define QK_SCALE 0.08838834764831845f  // 1/sqrt(128)

// Async copy of one 16-byte chunk global->LDS (CDNA5, tracked by ASYNCcnt).
// VDST operand = LDS byte address (low 32 bits of the generic pointer).
__device__ __forceinline__ void async_ld_b128(unsigned lds_off,
                                              unsigned long long gaddr) {
  asm volatile("global_load_async_to_lds_b128 %0, %1, off"
               :: "v"(lds_off), "v"(gaddr) : "memory");
}

// ---------------------------------------------------------------------------
// 1) Pack KAN weights: Wt[out][k], k<128 -> base_w; k>=128 -> spline_w*scaler.
// ---------------------------------------------------------------------------
__global__ void pack_weights(const float* __restrict__ base,
                             const float* __restrict__ spline,
                             const float* __restrict__ scaler,
                             _Float16* __restrict__ Wt, int out_dim) {
  int idx = blockIdx.x * blockDim.x + threadIdx.x;
  int total = out_dim * KDIM;
  if (idx >= total) return;
  int n = idx / KDIM;
  int k = idx % KDIM;
  float v;
  if (k < IN_DIM) {
    v = base[n * IN_DIM + k];
  } else {
    int km = k - IN_DIM;
    int i = km >> 3, c = km & 7;
    v = spline[((size_t)n * IN_DIM + i) * NCOEF + c] * scaler[n * IN_DIM + i];
  }
  Wt[idx] = (_Float16)v;
}

// ---------------------------------------------------------------------------
// 2) Featurize: phi[row][0:128] = silu(x), phi[row][128+i*8+c] = B-spline
//    bases (Cox-de Boor, order 3, uniform grid, h=0.4, 12 knots).
// ---------------------------------------------------------------------------
__global__ __launch_bounds__(128) void featurize(const float* __restrict__ X,
                                                 _Float16* __restrict__ phi) {
  int row = blockIdx.x;
  int i = threadIdx.x;
  float xv = X[(size_t)row * IN_DIM + i];
  float sil = xv / (1.0f + __expf(-xv));
  _Float16* prow = phi + (size_t)row * KDIM;
  prow[i] = (_Float16)sil;

  const float h = 0.4f;
  float g[12];
#pragma unroll
  for (int j = 0; j < 12; ++j) g[j] = -1.0f + (float)(j - 3) * h;
  float bb[11];
#pragma unroll
  for (int j = 0; j < 11; ++j)
    bb[j] = (xv >= g[j] && xv < g[j + 1]) ? 1.0f : 0.0f;
#pragma unroll
  for (int k = 1; k <= 3; ++k) {
#pragma unroll
    for (int j = 0; j < 11; ++j) {
      if (j <= 10 - k) {
        float left  = (xv - g[j]) / (g[j + k] - g[j]) * bb[j];
        float right = (g[j + k + 1] - xv) / (g[j + k + 1] - g[j + 1]) * bb[j + 1];
        bb[j] = left + right;
      }
    }
  }
#pragma unroll
  for (int c = 0; c < 8; ++c)
    prow[IN_DIM + i * NCOEF + c] = (_Float16)bb[c];
}

// ---------------------------------------------------------------------------
// 3) KAN GEMM: one wave computes a 16x64 strip (4 N-tiles share one A
//    fragment). K-loop unrolled by 64 with two disjoint fragment buffers so
//    loads for chunk k+32 overlap the WMMAs of chunk k (partial loadcnt
//    waits instead of wait-0 stalls).
// ---------------------------------------------------------------------------
__global__ __launch_bounds__(256) void kan_gemm(
    const _Float16* __restrict__ phi, const _Float16* __restrict__ Wt,
    int groupsN, int totalGroups, int mode,
    _Float16* __restrict__ Qb, _Float16* __restrict__ Kb,
    _Float16* __restrict__ Vt, float* __restrict__ Y) {
  int wave = threadIdx.x >> 5;
  int lane = threadIdx.x & 31;
  int grp = blockIdx.x * (blockDim.x >> 5) + wave;
  if (grp >= totalGroups) return;
  int gn = grp % groupsN, tm = grp / groupsN;
  int m0 = tm * 16, n0 = gn * 64;
  int half = lane >> 4, nl = lane & 15;

  const _Float16* arow = phi + (size_t)(m0 + nl) * KDIM + half * 16;
  const _Float16* brow0 = Wt + (size_t)(n0 + nl) * KDIM + half * 16;

  v8f acc[4];
#pragma unroll
  for (int j = 0; j < 4; ++j) acc[j] = (v8f){};

  ABFrag a0, a1, b0[4], b1[4];
  // preload K-chunk 0 into buffer 0
  a0.h[0] = *(const h8*)(arow);
  a0.h[1] = *(const h8*)(arow + 8);
#pragma unroll
  for (int j = 0; j < 4; ++j) {
    const _Float16* br = brow0 + (size_t)j * 16 * KDIM;
    b0[j].h[0] = *(const h8*)(br);
    b0[j].h[1] = *(const h8*)(br + 8);
  }

  for (int kk = 0; kk < KDIM; kk += 64) {
    int k1 = kk + 32;                                  // always < KDIM
    __builtin_prefetch(arow + kk + 256, 0, 1);
    // load buffer 1 (chunk k+32) while buffer 0 WMMAs run
    a1.h[0] = *(const h8*)(arow + k1);
    a1.h[1] = *(const h8*)(arow + k1 + 8);
#pragma unroll
    for (int j = 0; j < 4; ++j) {
      const _Float16* br = brow0 + (size_t)j * 16 * KDIM;
      b1[j].h[0] = *(const h8*)(br + k1);
      b1[j].h[1] = *(const h8*)(br + k1 + 8);
    }
#pragma unroll
    for (int j = 0; j < 4; ++j)
      acc[j] = __builtin_amdgcn_wmma_f32_16x16x32_f16(false, a0.v, false, b0[j].v,
                                                      (short)0, acc[j], false, false);
    int k2 = (kk + 64 < KDIM) ? kk + 64 : 0;           // clamp tail (redundant reload)
    a0.h[0] = *(const h8*)(arow + k2);
    a0.h[1] = *(const h8*)(arow + k2 + 8);
#pragma unroll
    for (int j = 0; j < 4; ++j) {
      const _Float16* br = brow0 + (size_t)j * 16 * KDIM;
      b0[j].h[0] = *(const h8*)(br + k2);
      b0[j].h[1] = *(const h8*)(br + k2 + 8);
    }
#pragma unroll
    for (int j = 0; j < 4; ++j)
      acc[j] = __builtin_amdgcn_wmma_f32_16x16x32_f16(false, a1.v, false, b1[j].v,
                                                      (short)0, acc[j], false, false);
  }

#pragma unroll
  for (int j = 0; j < 4; ++j) {
#pragma unroll
    for (int r = 0; r < 8; ++r) {
      int m = m0 + r + 8 * half;
      int nn = n0 + j * 16 + nl;
      float v = acc[j][r];
      if (mode == 0) {
        if (nn < 128) {
          Qb[(size_t)m * 128 + nn] = (_Float16)(v * QK_SCALE);
        } else if (nn < 256) {
          Kb[(size_t)m * 128 + (nn - 128)] = (_Float16)v;
        } else {
          int bb = m >> 11, t = m & 2047;
          Vt[((size_t)(bb * 128 + (nn - 256))) * NTOK + t] = (_Float16)v;
        }
      } else {
        Y[(size_t)m * 128 + nn] = v;
      }
    }
  }
}

// ---------------------------------------------------------------------------
// 4) Masked flash attention, 32 keys per iteration (no zero-padded WMMAs).
//    Adjacency tiles staged to LDS via global_load_async_to_lds_b128.
//    All B fragments for each WMMA chain are loaded up front so the loads
//    clause together and the chain runs behind one partial wait.
// ---------------------------------------------------------------------------
__global__ __launch_bounds__(128) void attn_kernel(
    const _Float16* __restrict__ Q, const _Float16* __restrict__ Kb,
    const _Float16* __restrict__ Vt, const int* __restrict__ adj,
    float* __restrict__ outp) {
  __shared__ __attribute__((aligned(32))) _Float16 plds[4][16 * 32]; // P tile
  __shared__ __attribute__((aligned(32))) int      alds[4][16 * 32]; // adj tile
  int wave = threadIdx.x >> 5;
  int lane = threadIdx.x & 31;
  int gt = blockIdx.x * 4 + wave;     // 0..511 q-tiles
  int b = gt >> 7;
  int m0 = (gt & 127) * 16;
  int half = lane >> 4, nl = lane & 15;

  // Q fragments (16x128, 4 K-chunks of 32)
  ABFrag aQ[4];
  const _Float16* qrow = Q + ((size_t)b * NTOK + m0 + nl) * 128;
#pragma unroll
  for (int c = 0; c < 4; ++c) {
    int kb = c * 32 + half * 16;
    aQ[c].h[0] = *(const h8*)(qrow + kb);
    aQ[c].h[1] = *(const h8*)(qrow + kb + 8);
  }

  v8f acc[8];
  float rmax[8], rsum[8];
#pragma unroll
  for (int nc = 0; nc < 8; ++nc) acc[nc] = (v8f){};
#pragma unroll
  for (int r = 0; r < 8; ++r) { rmax[r] = -3.0e38f; rsum[r] = 0.0f; }

  const char* adjb = (const char*)(adj + (size_t)b * NTOK * NTOK + (size_t)m0 * NTOK);
  unsigned abase = (unsigned)(uintptr_t)(&alds[wave][0]);

  for (int j0 = 0; j0 < NTOK; j0 += 32) {
    // ---- async-stage the 16x32 adjacency tile (2KB) into LDS ----
#pragma unroll
    for (int i = 0; i < 4; ++i) {
      int c = lane + 32 * i;                  // chunk 0..127, 16B each
      int row = c >> 3;                       // 8 chunks per 128B row
      int co = (c & 7) * 16;                  // byte offset within row
      unsigned long long g = (unsigned long long)(uintptr_t)
          (adjb + (size_t)row * (NTOK * 4) + (size_t)j0 * 4 + co);
      async_ld_b128(abase + (unsigned)c * 16, g);
    }

    // ---- scores: load all 8 K fragments, then two 4-deep WMMA chains ----
    ABFrag bk[8];
#pragma unroll
    for (int c = 0; c < 4; ++c) {
      const _Float16* kr0 = Kb + ((size_t)b * NTOK + j0 + nl) * 128 + c * 32 + half * 16;
      const _Float16* kr1 = kr0 + (size_t)16 * 128;
      bk[2 * c].h[0] = *(const h8*)(kr0);
      bk[2 * c].h[1] = *(const h8*)(kr0 + 8);
      bk[2 * c + 1].h[0] = *(const h8*)(kr1);
      bk[2 * c + 1].h[1] = *(const h8*)(kr1 + 8);
    }
    v8f s0 = {}, s1 = {};
#pragma unroll
    for (int c = 0; c < 4; ++c) {
      s0 = __builtin_amdgcn_wmma_f32_16x16x32_f16(false, aQ[c].v, false, bk[2 * c].v,
                                                  (short)0, s0, false, false);
      s1 = __builtin_amdgcn_wmma_f32_16x16x32_f16(false, aQ[c].v, false, bk[2 * c + 1].v,
                                                  (short)0, s1, false, false);
    }

    asm volatile("s_wait_asynccnt 0" ::: "memory");

    // ---- mask + online softmax over 32 columns ----
#pragma unroll
    for (int r = 0; r < 8; ++r) {
      int m = r + 8 * half;
      int a0 = alds[wave][m * 32 + nl];
      int a1 = alds[wave][m * 32 + 16 + nl];
      float sv0 = (a0 > 0) ? s0[r] : -3.0e38f;
      float sv1 = (a1 > 0) ? s1[r] : -3.0e38f;
      float tmax = fmaxf(sv0, sv1);
      tmax = fmaxf(tmax, __shfl_xor(tmax, 1, 32));
      tmax = fmaxf(tmax, __shfl_xor(tmax, 2, 32));
      tmax = fmaxf(tmax, __shfl_xor(tmax, 4, 32));
      tmax = fmaxf(tmax, __shfl_xor(tmax, 8, 32));
      float mnew = fmaxf(rmax[r], tmax);
      float scale = __expf(rmax[r] - mnew);
      float pv0 = __expf(sv0 - mnew);
      float pv1 = __expf(sv1 - mnew);
      float ps = pv0 + pv1;
      ps += __shfl_xor(ps, 1, 32);
      ps += __shfl_xor(ps, 2, 32);
      ps += __shfl_xor(ps, 4, 32);
      ps += __shfl_xor(ps, 8, 32);
      rsum[r] = rsum[r] * scale + ps;
      rmax[r] = mnew;
      plds[wave][m * 32 + nl] = (_Float16)pv0;
      plds[wave][m * 32 + 16 + nl] = (_Float16)pv1;
#pragma unroll
      for (int nc = 0; nc < 8; ++nc) acc[nc][r] *= scale;
    }

    // ---- C-layout P -> full 16x32 A fragment via LDS ----
    asm volatile("s_wait_dscnt 0" ::: "memory");
    ABFrag aP;
    aP.h[0] = *(const h8*)(&plds[wave][nl * 32 + half * 16]);
    aP.h[1] = *(const h8*)(&plds[wave][nl * 32 + half * 16 + 8]);

    // ---- O += P @ V_j : load all 8 V fragments, then 8 independent WMMAs ----
    ABFrag bv[8];
#pragma unroll
    for (int nc = 0; nc < 8; ++nc) {
      const _Float16* vrow = Vt + ((size_t)(b * 128 + nc * 16 + nl)) * NTOK
                             + j0 + 16 * half;
      bv[nc].h[0] = *(const h8*)(vrow);
      bv[nc].h[1] = *(const h8*)(vrow + 8);
    }
#pragma unroll
    for (int nc = 0; nc < 8; ++nc)
      acc[nc] = __builtin_amdgcn_wmma_f32_16x16x32_f16(false, aP.v, false, bv[nc].v,
                                                       (short)0, acc[nc], false, false);
  }

  // ---- normalize and write ----
#pragma unroll
  for (int r = 0; r < 8; ++r) {
    float inv = 1.0f / rsum[r];
    int m = m0 + r + 8 * half;
#pragma unroll
    for (int nc = 0; nc < 8; ++nc)
      outp[((size_t)b * NTOK + m) * 128 + nc * 16 + nl] = acc[nc][r] * inv;
  }
}

// ---------------------------------------------------------------------------
// 5) Epilogue: exact GELU then LayerNorm over the 128-dim row.
// ---------------------------------------------------------------------------
__global__ __launch_bounds__(128) void epilogue(const float* __restrict__ Yin,
                                                const float* __restrict__ gamma,
                                                const float* __restrict__ beta,
                                                float* __restrict__ outp) {
  __shared__ float red[128];
  int row = blockIdx.x, i = threadIdx.x;
  float y = Yin[(size_t)row * 128 + i];
  float gl = 0.5f * y * (1.0f + erff(y * 0.70710678118654752f));
  red[i] = gl;
  __syncthreads();
  for (int s = 64; s > 0; s >>= 1) {
    if (i < s) red[i] += red[i + s];
    __syncthreads();
  }
  float mu = red[0] * (1.0f / 128.0f);
  __syncthreads();
  float d = gl - mu;
  red[i] = d * d;
  __syncthreads();
  for (int s = 64; s > 0; s >>= 1) {
    if (i < s) red[i] += red[i + s];
    __syncthreads();
  }
  float var = red[0] * (1.0f / 128.0f);
  outp[(size_t)row * 128 + i] = d * rsqrtf(var + 1e-5f) * gamma[i] + beta[i];
}

// ---------------------------------------------------------------------------
// Launcher
// ---------------------------------------------------------------------------
extern "C" void kernel_launch(void* const* d_in, const int* in_sizes, int n_in,
                              void* d_out, int out_size, void* d_ws, size_t ws_size,
                              hipStream_t stream) {
  const float* x          = (const float*)d_in[0];
  const int*   adjacency  = (const int*)d_in[1];
  const float* q_base     = (const float*)d_in[2];
  const float* q_spline   = (const float*)d_in[3];
  const float* q_scaler   = (const float*)d_in[4];
  const float* k_base     = (const float*)d_in[5];
  const float* k_spline   = (const float*)d_in[6];
  const float* k_scaler   = (const float*)d_in[7];
  const float* v_base     = (const float*)d_in[8];
  const float* v_spline   = (const float*)d_in[9];
  const float* v_scaler   = (const float*)d_in[10];
  const float* o_base     = (const float*)d_in[11];
  const float* o_spline   = (const float*)d_in[12];
  const float* o_scaler   = (const float*)d_in[13];
  const float* ln_gamma   = (const float*)d_in[14];
  const float* ln_beta    = (const float*)d_in[15];
  float* out = (float*)d_out;

  // Workspace layout (256B-aligned offsets), ~34.7 MB total
  char* w = (char*)d_ws;
  _Float16* phi   = (_Float16*)(w + 0);                         // 8192*1152*2
  _Float16* WtQKV = (_Float16*)(w + 18874368);                  // 384*1152*2
  _Float16* WtO   = (_Float16*)(w + 19759104);                  // 128*1152*2
  _Float16* Qb    = (_Float16*)(w + 20054016);                  // 8192*128*2
  _Float16* Kb    = (_Float16*)(w + 22151168);                  // 8192*128*2
  _Float16* Vt    = (_Float16*)(w + 24248320);                  // 4*128*2048*2
  float*    attnO = (float*)   (w + 26345472);                  // 8192*128*4
  float*    Ybuf  = (float*)   (w + 30539776);                  // 8192*128*4
  (void)ws_size; (void)in_sizes; (void)n_in; (void)out_size;

  // 1) pack weights (q,k,v into one 384-row Wt; o separate)
  {
    int tot = 128 * KDIM;
    int blk = 256, grd = (tot + blk - 1) / blk;
    pack_weights<<<grd, blk, 0, stream>>>(q_base, q_spline, q_scaler, WtQKV, 128);
    pack_weights<<<grd, blk, 0, stream>>>(k_base, k_spline, k_scaler, WtQKV + (size_t)128 * KDIM, 128);
    pack_weights<<<grd, blk, 0, stream>>>(v_base, v_spline, v_scaler, WtQKV + (size_t)256 * KDIM, 128);
    pack_weights<<<grd, blk, 0, stream>>>(o_base, o_spline, o_scaler, WtO, 128);
  }
  // 2) featurize x
  featurize<<<ROWS, 128, 0, stream>>>(x, phi);
  // 3) QKV GEMM: 512 m-tiles x 6 N-groups (16x64 per wave), 8 waves/block
  {
    int groupsN = 6, total = (ROWS / 16) * groupsN;   // 3072
    kan_gemm<<<total / 8, 256, 0, stream>>>(phi, WtQKV, groupsN, total, 0,
                                            Qb, Kb, Vt, (float*)nullptr);
  }
  // 4) masked flash attention: 512 q-tiles, 4 waves/block
  attn_kernel<<<128, 128, 0, stream>>>(Qb, Kb, Vt, adjacency, attnO);
  // 5) featurize attention output (reuse phi)
  featurize<<<ROWS, 128, 0, stream>>>(attnO, phi);
  // 6) output KAN GEMM -> f32: 512 m-tiles x 2 N-groups
  {
    int groupsN = 2, total = (ROWS / 16) * groupsN;   // 1024
    kan_gemm<<<total / 8, 256, 0, stream>>>(phi, WtO, groupsN, total, 1,
                                            (_Float16*)nullptr, (_Float16*)nullptr,
                                            (_Float16*)nullptr, Ybuf);
  }
  // 7) GELU + LayerNorm
  epilogue<<<ROWS, 128, 0, stream>>>(Ybuf, ln_gamma, ln_beta, out);
}
